// GNNAttention_12970801234527
// MI455X (gfx1250) — matrix operand order repaired and checked by reading, not compiled, not run on hardware
//
#include <hip/hip_runtime.h>
#include <hip/hip_bf16.h>
#include <math.h>

// ---------------------------------------------------------------------------
// Problem constants (from reference): B=2, N_PTS=8192, D=128, K=16
// ---------------------------------------------------------------------------
namespace {
constexpr int BB = 2;
constexpr int NP = 8192;
constexpr int DD = 128;     // feature dim
constexpr int KK = 16;      // neighbors
constexpr int NT = BB * NP; // 16384 total nodes per cloud
}

typedef __attribute__((ext_vector_type(16))) __bf16 v16bf;
typedef __attribute__((ext_vector_type(8)))  float  v8f;

union FragB16 {
    v16bf h;
    uint4 q[2];
};

__device__ __forceinline__ unsigned short f2bf(float f) {
    union { float f; unsigned u; } v; v.f = f;
    unsigned r = v.u + 0x7FFFu + ((v.u >> 16) & 1u); // round-to-nearest-even
    return (unsigned short)(r >> 16);
}

// ---------------------------------------------------------------------------
// Utility: zero a float buffer (agg buffers + BN stats must start at 0)
// ---------------------------------------------------------------------------
__global__ void zero_f32_kernel(float* __restrict__ p, int n) {
    int t = blockIdx.x * blockDim.x + threadIdx.x;
    if (t < n) p[t] = 0.0f;
}

// ---------------------------------------------------------------------------
// Pack weight matrix W[256][128] (row-major, z-dim x out-dim) into a WMMA
// B-fragment-ready bf16 layout:
//   idx = ((kk*8 + nt)*32 + lane)*16 + j
//   lane holds column n = nt*16 + (lane&15), k = kk*32 + 8*(lane>>4) + j (j<8), kk*32 + 16 + 8*(lane>>4) + (j-8) (j>=8)
// so each lane's 16 bf16 (32B) are contiguous -> two b128 loads per fragment.
// ---------------------------------------------------------------------------
__global__ void pack_weights_kernel(const float* __restrict__ W,
                                    unsigned short* __restrict__ Wp) {
    int t = blockIdx.x * blockDim.x + threadIdx.x;
    if (t >= 8 * 8 * 32 * 16) return;
    int j    = t & 15;
    int lane = (t >> 4) & 31;
    int nt   = (t >> 9) & 7;
    int kk   = (t >> 12) & 7;
    int n = nt * 16 + (lane & 15);
    int k = kk * 32 + 8 * (lane >> 4) + (j < 8 ? j : 8 + j);
    Wp[t] = f2bf(W[k * DD + n]);
}

// ---------------------------------------------------------------------------
// Self kNN (euclidean, exclude self). One thread per query point, candidate
// tiles staged through LDS. Writes GLOBAL node ids (batch offset included).
// ---------------------------------------------------------------------------
__global__ void __launch_bounds__(256)
knn_self_kernel(const float* __restrict__ xyz, int* __restrict__ idx_out) {
    const int b = blockIdx.y;
    const int q = blockIdx.x * blockDim.x + threadIdx.x; // 0..NP-1
    const float* base = xyz + (size_t)b * NP * 3;
    const float qx = base[q * 3 + 0];
    const float qy = base[q * 3 + 1];
    const float qz = base[q * 3 + 2];

    __shared__ float sx[256], sy[256], sz[256];

    float tv[KK]; int ti[KK];
    #pragma unroll
    for (int j = 0; j < KK; ++j) { tv[j] = 1e30f; ti[j] = 0; }
    float worst = 1e30f; int wslot = 0;

    for (int t0 = 0; t0 < NP; t0 += 256) {
        int j = t0 + threadIdx.x;
        sx[threadIdx.x] = base[j * 3 + 0];
        sy[threadIdx.x] = base[j * 3 + 1];
        sz[threadIdx.x] = base[j * 3 + 2];
        __syncthreads();
        for (int jj = 0; jj < 256; ++jj) {
            int cand = t0 + jj;
            float dx = qx - sx[jj], dy = qy - sy[jj], dz = qz - sz[jj];
            float d2 = dx * dx + dy * dy + dz * dz;
            if (cand != q && d2 < worst) {
                tv[wslot] = d2; ti[wslot] = cand;
                worst = tv[0]; wslot = 0;
                #pragma unroll
                for (int s = 1; s < KK; ++s)
                    if (tv[s] > worst) { worst = tv[s]; wslot = s; }
            }
        }
        __syncthreads();
    }
    int qg = b * NP + q;
    #pragma unroll
    for (int j = 0; j < KK; ++j) idx_out[qg * KK + j] = b * NP + ti[j];
}

// ---------------------------------------------------------------------------
// CGConv message + aggregation (WMMA core).
//   Per dst node: A = z (16 edges x 256, bf16 in LDS), W packed bf16.
//   msg = sigmoid(z@Wf + bf) * softplus(z@Ws + bs)     (16 x 128, f32)
//   conv1 (scatter_base < 0): per-column max over the 16 edges -> direct store
//   conv2 (scatter_base >= 0): per-edge atomic float-max scatter (msg > 0)
// One wave per node; 4 waves / block; 32 KB LDS / block for z staging.
// ---------------------------------------------------------------------------
__global__ void __launch_bounds__(128)
cgconv_msg_kernel(const float* __restrict__ xc,   // center rows [.][D]
                  const float* __restrict__ xn,   // neighbor rows [.][D]
                  const int* __restrict__ nbr,    // [nodes][K] row ids into xn
                  const unsigned short* __restrict__ Wfp,
                  const unsigned short* __restrict__ Wsp,
                  const float* __restrict__ biasf,
                  const float* __restrict__ biass,
                  float* __restrict__ agg,
                  int agg_base, int center_first, int scatter_base) {
    __shared__ __align__(16) unsigned short zsh[4][16][256];
    __shared__ int nsh[4][16];

    const int wave = threadIdx.x >> 5;
    const int lane = threadIdx.x & 31;
    const int node = blockIdx.x * 4 + wave;

    if (lane < KK) nsh[wave][lane] = nbr[node * KK + lane];
    __syncthreads();

    // Stage z = [center, nbr] (conv1) or [nbr(dst), center(src)] (conv2), bf16
    for (int t = lane; t < 16 * 256; t += 32) {
        int e = t >> 8, c = t & 255;
        int isCenter = ((c < 128) == (center_first != 0));
        float v = isCenter ? xc[(size_t)node * DD + (c & 127)]
                           : xn[(size_t)nsh[wave][e] * DD + (c & 127)];
        zsh[wave][e][c] = f2bf(v);
    }
    __syncthreads();

    const int row = lane & 15, h = lane >> 4;
    for (int nt = 0; nt < 8; ++nt) {            // 8 tiles of 16 out channels
        v8f cf = {}; v8f cs = {};
        #pragma unroll
        for (int kk = 0; kk < 8; ++kk) {        // 8 k-steps of 32
            FragB16 a, wf, wsn;
            const unsigned short* zp = &zsh[wave][row][kk * 32 + 8 * h];
            a.q[0] = *(const uint4*)(zp);
            a.q[1] = *(const uint4*)(zp + 16);
            const uint4* wp = (const uint4*)(Wfp + (size_t)((kk * 8 + nt) * 32 + lane) * 16);
            wf.q[0] = wp[0]; wf.q[1] = wp[1];
            const uint4* sp = (const uint4*)(Wsp + (size_t)((kk * 8 + nt) * 32 + lane) * 16);
            wsn.q[0] = sp[0]; wsn.q[1] = sp[1];
            cf = __builtin_amdgcn_wmma_f32_16x16x32_bf16(false, a.h, false, wf.h,
                                                         (short)0, cf, false, false);
            cs = __builtin_amdgcn_wmma_f32_16x16x32_bf16(false, a.h, false, wsn.h,
                                                         (short)0, cs, false, false);
        }
        const int col = nt * 16 + row;          // lane's output channel
        const float bfv = biasf[col], bsv = biass[col];
        float m[8];
        #pragma unroll
        for (int r = 0; r < 8; ++r) {           // C row = edge e = r + 8h
            float f = cf[r] + bfv;
            float s = cs[r] + bsv;
            float sig = 1.0f / (1.0f + __expf(-f));
            float sp2 = (s > 20.0f) ? s : log1pf(__expf(s));
            m[r] = sig * sp2;                   // strictly > 0
        }
        if (scatter_base < 0) {
            // segment-max == max over this node's 16 edges, per column
            float mx = m[0];
            #pragma unroll
            for (int r = 1; r < 8; ++r) mx = fmaxf(mx, m[r]);
            mx = fmaxf(mx, __shfl_xor(mx, 16)); // combine row halves
            if (h == 0) agg[(size_t)(agg_base + node) * DD + col] = mx;
        } else {
            #pragma unroll
            for (int r = 0; r < 8; ++r) {
                const int e = r + 8 * h;
                const int drow = scatter_base + nsh[wave][e];
                // positive floats: int ordering == IEEE ordering
                atomicMax((int*)&agg[(size_t)drow * DD + col], __float_as_int(m[r]));
            }
        }
    }
}

// ---------------------------------------------------------------------------
// Row-normalize (cosine) f32 -> bf16. One wave per row.
// ---------------------------------------------------------------------------
__global__ void __launch_bounds__(256)
normalize_kernel(const float* __restrict__ x, unsigned short* __restrict__ out) {
    const int row  = blockIdx.x * 8 + (threadIdx.x >> 5);
    const int lane = threadIdx.x & 31;
    const float* xr = x + (size_t)row * DD;
    float v[4]; float ss = 0.0f;
    #pragma unroll
    for (int i = 0; i < 4; ++i) { v[i] = xr[lane + 32 * i]; ss += v[i] * v[i]; }
    #pragma unroll
    for (int o = 16; o > 0; o >>= 1) ss += __shfl_xor(ss, o);
    float r = rsqrtf(ss + 1e-12f);
    #pragma unroll
    for (int i = 0; i < 4; ++i) out[(size_t)row * DD + lane + 32 * i] = f2bf(v[i] * r);
}

// ---------------------------------------------------------------------------
// Cross cosine kNN: per 16-query strip (one wave), stream 16-candidate tiles,
// 16x16x128 GEMM per tile via 4 WMMA steps, threshold-gated top-16 per row.
// znorm layout: rows 0..NT-1 = f0c normalized, rows NT..2NT-1 = f1c normalized.
// Writes GLOBAL f1 node ids (0..NT-1).
// ---------------------------------------------------------------------------
__global__ void __launch_bounds__(32)
cosine_knn_kernel(const unsigned short* __restrict__ zn, int* __restrict__ idx_out) {
    const int lane  = threadIdx.x;
    const int b     = blockIdx.y;
    const int strip = blockIdx.x;              // 0..511
    const int q0    = b * NP + strip * 16;     // global f0 node id of row 0
    const unsigned short* qbase = zn;
    const unsigned short* cbase = zn + (size_t)NT * DD;

    const int row = lane & 15, h = lane >> 4;
    FragB16 a[4];
    #pragma unroll
    for (int kk = 0; kk < 4; ++kk) {           // A fragments stay in registers
        const unsigned short* p = qbase + (size_t)(q0 + row) * DD + kk * 32 + 8 * h;
        a[kk].q[0] = *(const uint4*)p;
        a[kk].q[1] = *(const uint4*)(p + 16);
    }

    float tv[KK]; int ti[KK];
    #pragma unroll
    for (int j = 0; j < KK; ++j) { tv[j] = -1e30f; ti[j] = 0; }
    float worst = -1e30f; int wslot = 0;

    __shared__ float S[16][16];

    for (int tile = 0; tile < NP / 16; ++tile) {
        const int c0 = b * NP + tile * 16;     // global f1 node id of col 0
        if (tile + 1 < NP / 16) {              // prefetch next candidate tile
            const unsigned short* np_ =
                cbase + (size_t)(c0 + 16 + row) * DD + 16 * h;
            __builtin_prefetch(np_, 0, 3);
        }
        v8f c = {};
        #pragma unroll
        for (int kk = 0; kk < 4; ++kk) {
            FragB16 bfr;
            const unsigned short* p = cbase + (size_t)(c0 + row) * DD + kk * 32 + 8 * h;
            bfr.q[0] = *(const uint4*)p;
            bfr.q[1] = *(const uint4*)(p + 16);
            c = __builtin_amdgcn_wmma_f32_16x16x32_bf16(false, a[kk].h, false, bfr.h,
                                                        (short)0, c, false, false);
        }
        #pragma unroll
        for (int r = 0; r < 8; ++r) S[r + 8 * h][row] = c[r];
        __syncthreads();
        if (lane < 16) {                       // lane owns query row `lane`
            for (int j = 0; j < 16; ++j) {
                float s = S[lane][j];
                if (s > worst) {
                    tv[wslot] = s; ti[wslot] = c0 + j;
                    worst = tv[0]; wslot = 0;
                    #pragma unroll
                    for (int t2 = 1; t2 < KK; ++t2)
                        if (tv[t2] < worst) { worst = tv[t2]; wslot = t2; }
                }
            }
        }
        __syncthreads();
    }
    if (lane < 16) {
        int q = q0 + lane;
        #pragma unroll
        for (int j = 0; j < KK; ++j) idx_out[q * KK + j] = ti[j];
    }
}

// ---------------------------------------------------------------------------
// BN batch stats: per-channel sum / sumsq via atomics into stats[0..127]/[128..255]
// ---------------------------------------------------------------------------
__global__ void reduce_stats_kernel(const float* __restrict__ agg, int rows,
                                    float* __restrict__ stats) {
    int tid = blockIdx.x * blockDim.x + threadIdx.x;
    int chan = tid & 127;
    int stride = (gridDim.x * blockDim.x) >> 7;
    float s = 0.0f, q = 0.0f;
    for (int row = tid >> 7; row < rows; row += stride) {
        float v = agg[(size_t)row * DD + chan];
        s += v; q += v * v;
    }
    atomicAdd(&stats[chan], s);
    atomicAdd(&stats[128 + chan], q);
}

// out = gamma*(agg-mean)*rsqrt(var+eps) + beta + x   (biased var, eps=1e-5)
__global__ void bn_apply_kernel(const float* __restrict__ agg,
                                const float* __restrict__ x,
                                const float* __restrict__ stats,
                                const float* __restrict__ g,
                                const float* __restrict__ be,
                                float* __restrict__ out, int rows, float inv_n) {
    int t = blockIdx.x * blockDim.x + threadIdx.x;
    if (t >= rows * DD) return;
    int c = t & 127;
    float mean = stats[c] * inv_n;
    float var  = stats[128 + c] * inv_n - mean * mean;
    out[t] = g[c] * (agg[t] - mean) * rsqrtf(var + 1e-5f) + be[c] + x[t];
}

// ---------------------------------------------------------------------------
// Host-side orchestration
// ---------------------------------------------------------------------------
extern "C" void kernel_launch(void* const* d_in, const int* in_sizes, int n_in,
                              void* d_out, int out_size, void* d_ws, size_t ws_size,
                              hipStream_t stream) {
    (void)in_sizes; (void)n_in; (void)out_size; (void)ws_size;

    const float* f0  = (const float*)d_in[2];
    const float* f1  = (const float*)d_in[3];
    const float* Wf1 = (const float*)d_in[4];
    const float* bf1 = (const float*)d_in[5];
    const float* Ws1 = (const float*)d_in[6];
    const float* bs1 = (const float*)d_in[7];
    const float* g1  = (const float*)d_in[8];
    const float* be1 = (const float*)d_in[9];
    const float* Wf2 = (const float*)d_in[10];
    const float* bf2 = (const float*)d_in[11];
    const float* Ws2 = (const float*)d_in[12];
    const float* bs2 = (const float*)d_in[13];
    const float* g2  = (const float*)d_in[14];
    const float* be2 = (const float*)d_in[15];
    float* out = (float*)d_out;

    char* ws = (char*)d_ws;
    size_t off = 0;
    auto alloc = [&](size_t bytes) -> void* {
        void* p = ws + off;
        off = (off + bytes + 255) & ~(size_t)255;
        return p;
    };
    float* fcat          = (float*)alloc((size_t)2 * NT * DD * 4); // [f0c | f1c]
    float* agg           = (float*)alloc((size_t)2 * NT * DD * 4);
    unsigned short* znrm = (unsigned short*)alloc((size_t)2 * NT * DD * 2);
    unsigned short* Wf1p = (unsigned short*)alloc(65536);
    unsigned short* Ws1p = (unsigned short*)alloc(65536);
    unsigned short* Wf2p = (unsigned short*)alloc(65536);
    unsigned short* Ws2p = (unsigned short*)alloc(65536);
    int* idx0            = (int*)alloc((size_t)NT * KK * 4);
    int* idx1            = (int*)alloc((size_t)NT * KK * 4);
    int* idxc            = (int*)alloc((size_t)NT * KK * 4);
    float* stats         = (float*)alloc(768 * 4);  // 3 x (sum|sumsq) of 128

    const int aggN = 2 * NT * DD;
    // 0) zero scratch that must start from 0 every call (graph-replay safe)
    zero_f32_kernel<<<(aggN + 255) / 256, 256, 0, stream>>>(agg, aggN);
    zero_f32_kernel<<<3, 256, 0, stream>>>(stats, 768);

    // 1) pack weights into WMMA B-fragment layout (bf16)
    pack_weights_kernel<<<128, 256, 0, stream>>>(Wf1, Wf1p);
    pack_weights_kernel<<<128, 256, 0, stream>>>(Ws1, Ws1p);
    pack_weights_kernel<<<128, 256, 0, stream>>>(Wf2, Wf2p);
    pack_weights_kernel<<<128, 256, 0, stream>>>(Ws2, Ws2p);

    // 2) self kNN graphs on coordinates
    knn_self_kernel<<<dim3(NP / 256, BB), 256, 0, stream>>>((const float*)d_in[0], idx0);
    knn_self_kernel<<<dim3(NP / 256, BB), 256, 0, stream>>>((const float*)d_in[1], idx1);

    // 3) conv1 messages + per-node max (shared weights, both clouds)
    cgconv_msg_kernel<<<NT / 4, 128, 0, stream>>>(f0, f0, idx0, Wf1p, Ws1p, bf1, bs1,
                                                  agg, 0, 1, -1);
    cgconv_msg_kernel<<<NT / 4, 128, 0, stream>>>(f1, f1, idx1, Wf1p, Ws1p, bf1, bs1,
                                                  agg, NT, 1, -1);

    // 4) conv1 BN stats + apply (residual): fcat = BN(agg)+x
    reduce_stats_kernel<<<512, 256, 0, stream>>>(agg, NT, stats);
    reduce_stats_kernel<<<512, 256, 0, stream>>>(agg + (size_t)NT * DD, NT, stats + 256);
    bn_apply_kernel<<<(NT * DD + 255) / 256, 256, 0, stream>>>(
        agg, f0, stats, g1, be1, fcat, NT, 1.0f / NT);
    bn_apply_kernel<<<(NT * DD + 255) / 256, 256, 0, stream>>>(
        agg + (size_t)NT * DD, f1, stats + 256, g1, be1, fcat + (size_t)NT * DD,
        NT, 1.0f / NT);

    // 5) re-zero agg for conv2 scatter-max (f0 half stays 0 = empty segments)
    zero_f32_kernel<<<(aggN + 255) / 256, 256, 0, stream>>>(agg, aggN);

    // 6) cosine cross kNN: normalize then WMMA similarity + streaming top-16
    normalize_kernel<<<2 * NT / 8, 256, 0, stream>>>(fcat, znrm);
    cosine_knn_kernel<<<dim3(NP / 16, BB), 32, 0, stream>>>(znrm, idxc);

    // 7) conv2 messages (z=[f1c[dst], f0c[src]]) + atomic scatter-max on dst
    cgconv_msg_kernel<<<NT / 4, 128, 0, stream>>>(fcat, fcat + (size_t)NT * DD, idxc,
                                                  Wf2p, Ws2p, bf2, bs2,
                                                  agg, 0, 0, NT);

    // 8) conv2 BN over all 2N rows (incl. zero f0 half) + residual -> d_out
    reduce_stats_kernel<<<512, 256, 0, stream>>>(agg, 2 * NT, stats + 512);
    bn_apply_kernel<<<(2 * NT * DD + 255) / 256, 256, 0, stream>>>(
        agg, fcat, stats + 512, g2, be2, out, 2 * NT, 1.0f / (2 * NT));
}
